// GraphAttention_52166672777276
// MI455X (gfx1250) — matrix operand order repaired and checked by reading, not compiled, not run on hardware
//
#include <hip/hip_runtime.h>
#include <stdint.h>

#define NN 50000      // nodes
#define NE 800000     // edges
#define DI 256        // d_in
#define DO 64         // d_out
#define NH 4          // heads
#define MT ((NN + 15) / 16)   // 3125 m-tiles, exact

typedef __attribute__((ext_vector_type(16))) __bf16 v16bf;
typedef __attribute__((ext_vector_type(8)))  __bf16 v8bf;
typedef __attribute__((ext_vector_type(8)))  float  v8f;

__device__ __forceinline__ unsigned short f2bf(float f) {
    unsigned u = __float_as_uint(f);
    u += 0x7FFFu + ((u >> 16) & 1u);   // round-to-nearest-even
    return (unsigned short)(u >> 16);
}

// float atomic max via int punning (works across sign mix; segmax init = -inf)
__device__ __forceinline__ void atomicMaxF(float* a, float v) {
    if (v >= 0.0f) atomicMax((int*)a, __float_as_int(v));
    else           atomicMin((unsigned int*)a, (unsigned int)__float_as_int(v));
}

// ---------- init: out=0, segmax=-inf, segsum=0 ----------
__global__ void k_init(float* __restrict__ out, float* __restrict__ segmax,
                       float* __restrict__ segsum) {
    size_t i = (size_t)blockIdx.x * blockDim.x + threadIdx.x;
    if (i < (size_t)NN * NH * DO) out[i] = 0.0f;
    if (i < (size_t)NN * NH) { segmax[i] = -__builtin_inff(); segsum[i] = 0.0f; }
}

// ---------- x (f32) -> bf16 ----------
__global__ void k_cvt_x(const float* __restrict__ x, unsigned short* __restrict__ xb) {
    size_t i = (size_t)blockIdx.x * blockDim.x + threadIdx.x;
    if (i < (size_t)NN * DI) xb[i] = f2bf(x[i]);
}

// ---------- pack W into per-lane WMMA B fragments ----------
// B frag order used here (32x16 KxN, contiguous k, same as A): lane n in [0,16): column N=n, K=2v,2v+1
// per VGPR v (K 0..15); lanes [16,32): same column, K 16..31. Stored 16 bf16/lane,
// fragment index = ((head*8 + ktile)*4 + ntile)*32 + lane.
__global__ void k_pack_w(const float* __restrict__ W, unsigned short* __restrict__ wf) {
    int i = blockIdx.x * blockDim.x + threadIdx.x;
    if (i >= NH * 8 * 4 * 32) return;
    int lane = i & 31;
    int nt   = (i >> 5) & 3;
    int kt   = (i >> 7) & 7;
    int h    = i >> 10;
    int kbase = kt * 32 + ((lane & 16) ? 16 : 0);
    int n     = nt * 16 + (lane & 15);
    unsigned short* dst = wf + (size_t)i * 16;
#pragma unroll
    for (int v = 0; v < 8; ++v) {
        dst[2*v]   = f2bf(W[((size_t)h * DI + kbase + 2*v    ) * DO + n]);
        dst[2*v+1] = f2bf(W[((size_t)h * DI + kbase + 2*v + 1) * DO + n]);
    }
}

// ---------- h[head][n][o] = x @ W[head], bf16 WMMA, f32 accum ----------
__global__ void __launch_bounds__(128) k_gemm(const unsigned short* __restrict__ xb,
                                              const unsigned short* __restrict__ wf,
                                              float* __restrict__ hbuf) {
    int wid = blockIdx.x * 4 + (threadIdx.x >> 5);
    if (wid >= NH * MT) return;
    int lane = threadIdx.x & 31;
    int head = wid / MT;
    int m0   = (wid % MT) * 16;

    // A frag (16x32 bf16): lanes<16 -> row m0+lane, K {0..15};
    // lanes>=16 -> row m0+lane-16, K {16..31}
    const __bf16* xrow  = (const __bf16*)xb + (size_t)(m0 + (lane & 15)) * DI;
    const int     koffA = (lane & 16) ? 16 : 0;
    const __bf16* wfb   = (const __bf16*)wf;

    v8f acc0 = {}, acc1 = {}, acc2 = {}, acc3 = {};
#pragma unroll
    for (int kt = 0; kt < 8; ++kt) {
        int kk = kt * 32;
        v8bf lo = *(const v8bf*)(xrow + kk + koffA);
        v8bf hi = *(const v8bf*)(xrow + kk + koffA + 8);
        v16bf afrag = __builtin_shufflevector(lo, hi, 0,1,2,3,4,5,6,7,8,9,10,11,12,13,14,15);

        const __bf16* bbase = wfb + (size_t)((head * 8 + kt) * 4) * 512 + (size_t)lane * 16;
        v16bf b0 = *(const v16bf*)(bbase + 0 * 512);
        v16bf b1 = *(const v16bf*)(bbase + 1 * 512);
        v16bf b2 = *(const v16bf*)(bbase + 2 * 512);
        v16bf b3 = *(const v16bf*)(bbase + 3 * 512);

        acc0 = __builtin_amdgcn_wmma_f32_16x16x32_bf16(false, afrag, false, b0, (short)0, acc0, false, false);
        acc1 = __builtin_amdgcn_wmma_f32_16x16x32_bf16(false, afrag, false, b1, (short)0, acc1, false, false);
        acc2 = __builtin_amdgcn_wmma_f32_16x16x32_bf16(false, afrag, false, b2, (short)0, acc2, false, false);
        acc3 = __builtin_amdgcn_wmma_f32_16x16x32_bf16(false, afrag, false, b3, (short)0, acc3, false, false);
    }

    // C layout: lane = column N (mod 16); VGPR r = row r (lanes<16) / r+8 (lanes>=16)
    float* hp = hbuf + ((size_t)head * NN + m0) * DO;
    int n  = lane & 15;
    int mb = (lane & 16) ? 8 : 0;
#pragma unroll
    for (int r = 0; r < 8; ++r) {
        hp[(mb + r) * DO +  0 + n] = acc0[r];
        hp[(mb + r) * DO + 16 + n] = acc1[r];
        hp[(mb + r) * DO + 32 + n] = acc2[r];
        hp[(mb + r) * DO + 48 + n] = acc3[r];
    }
}

// ---------- s1/s2: per (head,node) dots with a1/a2 ----------
__global__ void k_scores(const float* __restrict__ hbuf, const float* __restrict__ a,
                         float* __restrict__ s1, float* __restrict__ s2) {
    int i = blockIdx.x * blockDim.x + threadIdx.x;
    if (i >= NH * NN) return;
    int head = i / NN, node = i % NN;
    const float* hv = hbuf + ((size_t)head * NN + node) * DO;
    const float* a1 = a + head * 2 * DO;
    const float* a2 = a1 + DO;
    float r1 = 0.0f, r2 = 0.0f;
#pragma unroll 8
    for (int o = 0; o < DO; ++o) { float v = hv[o]; r1 += v * a1[o]; r2 += v * a2[o]; }
    s1[i] = r1; s2[i] = r2;
}

// ---------- edge pass 1: leaky-relu logit + segment max ----------
__global__ void k_edge_max(const int* __restrict__ ei, const float* __restrict__ s1,
                           const float* __restrict__ s2, float* __restrict__ logits,
                           float* __restrict__ segmax) {
    int i = blockIdx.x * blockDim.x + threadIdx.x;
    if (i >= NH * NE) return;
    int head = i / NE, e = i % NE;
    int row = ei[2 * e], col = ei[2 * e + 1];
    float v = s1[head * NN + row] + s2[head * NN + col];
    v = (v > 0.0f) ? v : 0.2f * v;
    logits[i] = v;
    atomicMaxF(&segmax[head * NN + row], v);
}

// ---------- edge pass 2: exp(logit - max) + segment sum ----------
__global__ void k_edge_exp(const int* __restrict__ ei, float* __restrict__ logits,
                           const float* __restrict__ segmax, float* __restrict__ segsum) {
    int i = blockIdx.x * blockDim.x + threadIdx.x;
    if (i >= NH * NE) return;
    int head = i / NE, e = i % NE;
    int row = ei[2 * e];
    float ex = __expf(logits[i] - segmax[head * NN + row]);
    logits[i] = ex;
    atomicAdd(&segsum[head * NN + row], ex);
}

// ---------- edge pass 3: alpha = ex / segsum[row] ----------
__global__ void k_alpha(const int* __restrict__ ei, float* __restrict__ logits,
                        const float* __restrict__ segsum) {
    int i = blockIdx.x * blockDim.x + threadIdx.x;
    if (i >= NH * NE) return;
    int head = i / NE, e = i % NE;
    int row = ei[2 * e];
    logits[i] = logits[i] / segsum[head * NN + row];
}

// ---------- edge pass 4: out[row, h, o] += alpha * h[h, col, o] ----------
__global__ void k_scatter(const int* __restrict__ ei, const float* __restrict__ alpha,
                          const float* __restrict__ hbuf, float* __restrict__ out) {
    size_t i = (size_t)blockIdx.x * blockDim.x + threadIdx.x;
    if (i >= (size_t)NE * NH * DO) return;
    int e = (int)(i >> 8);
    int f = (int)(i & 255);
    int head = f >> 6;
    int o    = f & 63;
    int row = ei[2 * e], col = ei[2 * e + 1];
    float al = alpha[(size_t)head * NE + e];
    float v  = al * hbuf[((size_t)head * NN + col) * DO + o];
    atomicAdd(out + (size_t)row * (NH * DO) + f, v);
}

// ---------- final relu in place ----------
__global__ void k_relu(float* __restrict__ out) {
    size_t i = (size_t)blockIdx.x * blockDim.x + threadIdx.x;
    if (i < (size_t)NN * NH * DO) out[i] = fmaxf(out[i], 0.0f);
}

extern "C" void kernel_launch(void* const* d_in, const int* in_sizes, int n_in,
                              void* d_out, int out_size, void* d_ws, size_t ws_size,
                              hipStream_t stream) {
    const float* x  = (const float*)d_in[0];   // (50000, 256)
    const float* W  = (const float*)d_in[1];   // (4, 256, 64)
    const float* a  = (const float*)d_in[2];   // (4, 128)
    const int*   ei = (const int*)d_in[3];     // (800000, 2)
    float* out = (float*)d_out;                // (50000, 256)

    char* ws = (char*)d_ws;
    size_t off = 0;
    auto carve = [&](size_t bytes) { char* p = ws + off; off += (bytes + 255) & ~(size_t)255; return p; };

    unsigned short* xb     = (unsigned short*)carve((size_t)NN * DI * 2);      // 25.6 MB
    unsigned short* wf     = (unsigned short*)carve((size_t)NH * 8 * 4 * 32 * 16 * 2); // 128 KB
    float*          hbuf   = (float*)carve((size_t)NH * NN * DO * 4);          // 51.2 MB
    float*          s1     = (float*)carve((size_t)NH * NN * 4);
    float*          s2     = (float*)carve((size_t)NH * NN * 4);
    float*          segmax = (float*)carve((size_t)NH * NN * 4);
    float*          segsum = (float*)carve((size_t)NH * NN * 4);
    float*          logits = (float*)carve((size_t)NH * NE * 4);               // 12.8 MB

    const int B = 256;
    k_init   <<<((size_t)NN * NH * DO + B - 1) / B, B, 0, stream>>>(out, segmax, segsum);
    k_cvt_x  <<<((size_t)NN * DI + B - 1) / B,      B, 0, stream>>>(x, xb);
    k_pack_w <<<(NH * 8 * 4 * 32 + B - 1) / B,      B, 0, stream>>>(W, wf);
    k_gemm   <<<(NH * MT + 3) / 4,                128, 0, stream>>>(xb, wf, hbuf);
    k_scores <<<(NH * NN + B - 1) / B,              B, 0, stream>>>(hbuf, a, s1, s2);
    k_edge_max<<<(NH * NE + B - 1) / B,             B, 0, stream>>>(ei, s1, s2, logits, segmax);
    k_edge_exp<<<(NH * NE + B - 1) / B,             B, 0, stream>>>(ei, logits, segmax, segsum);
    k_alpha  <<<(NH * NE + B - 1) / B,              B, 0, stream>>>(ei, logits, segsum);
    k_scatter<<<(unsigned)(((size_t)NE * NH * DO + B - 1) / B), B, 0, stream>>>(ei, logits, hbuf, out);
    k_relu   <<<((size_t)NN * NH * DO + B - 1) / B, B, 0, stream>>>(out);
}
